// MappingNetwork_52621939310621
// MI455X (gfx1250) — matrix operand order repaired and checked by reading, not compiled, not run on hardware
//
#include <hip/hip_runtime.h>

#define HID   256
#define STYLE 64
#define NSPK  10
#define AROW  264   // LDS activation row stride in halfs (256 + 8 pad)
#define WROW  40    // LDS weight-chunk row stride in halfs (32 + 8 pad)
#define TPB   128   // 4 waves of 32

typedef __attribute__((ext_vector_type(16))) _Float16 v16h;
typedef __attribute__((ext_vector_type(8)))  _Float16 v8h;
typedef __attribute__((ext_vector_type(8)))  float    v8f;

// Low 32 bits of a generic pointer to LDS == wave-relative LDS byte address
// (ISA 10.2).
__device__ __forceinline__ unsigned int lds_addr32(const void* p) {
  return (unsigned int)(unsigned long long)p;
}

// Async DMA: 16 bytes global -> LDS per lane, tracked by ASYNCcnt.
__device__ __forceinline__ void async_b128(unsigned int lds, const void* gptr) {
  asm volatile("global_load_async_to_lds_b128 %0, %1, off"
               :: "v"(lds), "v"(gptr) : "memory");
}
__device__ __forceinline__ void wait_async0() {
  asm volatile("s_wait_asynccnt 0x0" ::: "memory");
}

// Issue one 32-wide K chunk of transposed weights Wt[N][K] into an LDS buffer.
template<int K, int N>
__device__ __forceinline__ void issue_chunk(_Float16* __restrict__ buf,
                                            const _Float16* __restrict__ wt,
                                            int tid, int kc) {
  const _Float16* src = wt + kc * 32;
  #pragma unroll
  for (int u = tid; u < N * 4; u += TPB) {
    int row = u >> 2, seg = u & 3;
    async_b128(lds_addr32(buf + row * WROW + seg * 8), src + row * K + seg * 8);
  }
}

// ---------------------------------------------------------------------------
// One dense layer: act[64][K] (f16, LDS) x Wt[N][K] (f16, global, transposed)
// -> act[64][N] or out rows.  4 waves x 16-row M tiles; weight K-chunks are
// double-buffered in LDS via async global->LDS DMA overlapped with WMMA.
// ---------------------------------------------------------------------------
template<int K, int N, bool RELU, bool TOACT>
__device__ __forceinline__ void mlp_layer(
    _Float16* __restrict__ act, _Float16* __restrict__ wb0, _Float16* __restrict__ wb1,
    const _Float16* __restrict__ wt, const float* __restrict__ bias,
    int tid, float* __restrict__ outp, const int* __restrict__ bidx, int nvalid)
{
  constexpr int NK = K / 32;   // K chunks
  constexpr int NT = N / 16;   // N tiles
  const int lane = tid & 31;
  const int wv   = tid >> 5;
  const int kh   = lane >> 4;
  const int lrow = lane & 15;

  v8f acc[NT];
  v8f vzero = {};
  #pragma unroll
  for (int i = 0; i < NT; ++i) acc[i] = vzero;

  __syncthreads();                       // both chunk buffers free
  issue_chunk<K, N>(wb0, wt, tid, 0);    // prefetch chunk 0

  for (int kc = 0; kc < NK; ++kc) {
    wait_async0();      // my chunk-kc copies have landed (async loads retire in order)
    __syncthreads();    // everyone's copies visible; everyone done with other buffer
    if (kc + 1 < NK)    // DMA next chunk while we do the WMMAs below
      issue_chunk<K, N>((kc & 1) ? wb0 : wb1, wt, tid, kc + 1);
    const _Float16* cb = (kc & 1) ? wb1 : wb0;

    // A fragment (16x32 f16): lanes 0-15 row M=lrow hold K {0..7, 16..23},
    // lanes 16-31 hold K {8..15, 24..31}  (ISA 7.12.2 layout)
    const _Float16* ap = act + (wv * 16 + lrow) * AROW + kc * 32 + kh * 8;
    v8h alo = *reinterpret_cast<const v8h*>(ap);
    v8h ahi = *reinterpret_cast<const v8h*>(ap + 16);
    v16h a;
    #pragma unroll
    for (int i = 0; i < 8; ++i) { a[i] = alo[i]; a[i + 8] = ahi[i]; }

    #pragma unroll
    for (int nt = 0; nt < NT; ++nt) {
      // B fragment (32x16 f16): lane = column n, lanes 0-15 hold K {0..7, 16..23}, lanes 16-31 hold K {8..15, 24..31}
      const _Float16* bp = cb + (nt * 16 + lrow) * WROW + kh * 8;
      v8h blo = *reinterpret_cast<const v8h*>(bp);
      v8h bhi = *reinterpret_cast<const v8h*>(bp + 16);
      v16h bm;
      #pragma unroll
      for (int i = 0; i < 8; ++i) { bm[i] = blo[i]; bm[i + 8] = bhi[i]; }
      acc[nt] = __builtin_amdgcn_wmma_f32_16x16x32_f16(
          false, a, false, bm, (short)0, acc[nt], false, false);
    }
  }

  // Epilogue: bias + (ReLU) + store.  C/D layout: lane<16 elem e -> (M=e, N=lane),
  // lane>=16 elem e -> (M=8+e, N=lane-16).
  #pragma unroll
  for (int nt = 0; nt < NT; ++nt) {
    int n = nt * 16 + lrow;
    float bv = bias[n];
    #pragma unroll
    for (int e = 0; e < 8; ++e) {
      float v = acc[nt][e] + bv;
      if (RELU) v = fmaxf(v, 0.0f);
      int row = wv * 16 + kh * 8 + e;
      if (TOACT) {
        act[row * AROW + n] = (_Float16)v;       // own rows only; no barrier needed
      } else {
        if (row < nvalid) outp[bidx[row] * N + n] = v;
      }
    }
  }
}

// ---------------------------------------------------------------------------
// Shared trunk: z[B,16] -> h[B,256] (f16)
// ---------------------------------------------------------------------------
__global__ void __launch_bounds__(TPB)
trunk_kernel(const float* __restrict__ z,
             const _Float16* __restrict__ wt0, const _Float16* __restrict__ wt1,
             const _Float16* __restrict__ wt2, const _Float16* __restrict__ wt3,
             const float* __restrict__ sb0, const float* __restrict__ sb1,
             const float* __restrict__ sb2, const float* __restrict__ sb3,
             _Float16* __restrict__ h16, int Bn)
{
  __shared__ __align__(16) _Float16 act[64 * AROW];
  __shared__ __align__(16) _Float16 wb0[256 * WROW];
  __shared__ __align__(16) _Float16 wb1[256 * WROW];
  const int tid  = threadIdx.x;
  const int base = blockIdx.x * 64;

  // Load z tile, K padded 16->32 with zeros (wt0 rows 16..31 are also zero)
  for (int i = tid; i < 64 * 32; i += TPB) {
    int r = i >> 5, k = i & 31;
    int gr = base + r;
    float v = (k < 16 && gr < Bn) ? z[gr * 16 + k] : 0.0f;
    act[r * AROW + k] = (_Float16)v;
  }
  // (barriers inside mlp_layer cover the fill)

  mlp_layer< 32, HID, true, true>(act, wb0, wb1, wt0, sb0, tid, nullptr, nullptr, 0);
  mlp_layer<HID, HID, true, true>(act, wb0, wb1, wt1, sb1, tid, nullptr, nullptr, 0);
  mlp_layer<HID, HID, true, true>(act, wb0, wb1, wt2, sb2, tid, nullptr, nullptr, 0);
  mlp_layer<HID, HID, true, true>(act, wb0, wb1, wt3, sb3, tid, nullptr, nullptr, 0);
  __syncthreads();

  for (int i = tid; i < 64 * 32; i += TPB) {
    int r = i >> 5, seg = i & 31;
    int gr = base + r;
    if (gr < Bn)
      *reinterpret_cast<uint4*>(h16 + gr * HID + seg * 8) =
          *reinterpret_cast<const uint4*>(act + r * AROW + seg * 8);
  }
}

// ---------------------------------------------------------------------------
// Per-speaker branch: only the branch selected by y[b] is computed.
// One block = one 64-row tile of same-speaker samples.
// ---------------------------------------------------------------------------
__global__ void __launch_bounds__(TPB)
branch_kernel(const _Float16* __restrict__ h16,
              const _Float16* __restrict__ wu0, const _Float16* __restrict__ wu1,
              const _Float16* __restrict__ wu2, const _Float16* __restrict__ wu3,
              const float* __restrict__ ub0, const float* __restrict__ ub1,
              const float* __restrict__ ub2, const float* __restrict__ ub3,
              const int* __restrict__ idx,
              const int* __restrict__ tileSpk, const int* __restrict__ tileBase,
              const int* __restrict__ tileValid, const int* __restrict__ tcnt,
              float* __restrict__ out)
{
  __shared__ __align__(16) _Float16 act[64 * AROW];
  __shared__ __align__(16) _Float16 wb0[256 * WROW];
  __shared__ __align__(16) _Float16 wb1[256 * WROW];
  __shared__ int bidx[64];

  const int t = blockIdx.x;
  if (t >= *tcnt) return;           // uniform per block
  const int tid = threadIdx.x;
  const int s  = tileSpk[t];
  const int b0 = tileBase[t];
  const int nv = tileValid[t];

  if (tid < 64) {
    int j = tid < nv ? tid : nv - 1;  // pad rows replay a valid row; masked at store
    bidx[tid] = idx[b0 + j];
  }
  __syncthreads();

  // Async gather of this tile's h rows into LDS; drains at layer 0's asynccnt wait.
  for (int i = tid; i < 64 * 32; i += TPB) {
    int r = i >> 5, seg = i & 31;
    async_b128(lds_addr32(act + r * AROW + seg * 8), h16 + bidx[r] * HID + seg * 8);
  }

  mlp_layer<HID, HID,   true,  true >(act, wb0, wb1, wu0 + s * HID * HID,   ub0 + s * HID,   tid, nullptr, nullptr, 0);
  mlp_layer<HID, HID,   true,  true >(act, wb0, wb1, wu1 + s * HID * HID,   ub1 + s * HID,   tid, nullptr, nullptr, 0);
  mlp_layer<HID, HID,   true,  true >(act, wb0, wb1, wu2 + s * HID * HID,   ub2 + s * HID,   tid, nullptr, nullptr, 0);
  mlp_layer<HID, STYLE, false, false>(act, wb0, wb1, wu3 + s * STYLE * HID, ub3 + s * STYLE, tid, out, bidx, nv);
}

// ---------------------------------------------------------------------------
// Weight prep: f32 -> f16, transposed to [N][K] so the B fragment is contiguous
// ---------------------------------------------------------------------------
__global__ void conv_wt0(const float* __restrict__ src, _Float16* __restrict__ dst) {
  int i = blockIdx.x * 256 + threadIdx.x;      // [256 n][32 k], k 16..31 zero-pad
  if (i >= 256 * 32) return;
  int n = i >> 5, k = i & 31;
  dst[i] = (_Float16)((k < 16) ? src[k * 256 + n] : 0.0f);
}

__global__ void conv_sq(const float* __restrict__ src, _Float16* __restrict__ dst, int nmat) {
  int i = blockIdx.x * 256 + threadIdx.x;      // [m][256 n][256 k] <- [m][256 k][256 n]
  if (i >= nmat * 65536) return;
  int m = i >> 16, j = i & 65535;
  int n = j >> 8,  k = j & 255;
  dst[i] = (_Float16)src[m * 65536 + k * 256 + n];
}

__global__ void conv_wu3(const float* __restrict__ src, _Float16* __restrict__ dst) {
  int i = blockIdx.x * 256 + threadIdx.x;      // [s][64 n][256 k] <- [s][256 k][64 n]
  if (i >= NSPK * STYLE * HID) return;
  int s = i / (STYLE * HID), r = i % (STYLE * HID);
  int n = r >> 8, k = r & 255;
  dst[i] = (_Float16)src[s * STYLE * HID + k * STYLE + n];
}

// ---------------------------------------------------------------------------
// Speaker bucketing (output values are order-independent -> deterministic)
// ---------------------------------------------------------------------------
__global__ void bucket_zero(int* __restrict__ counts) {
  if (threadIdx.x < NSPK) counts[threadIdx.x] = 0;
}
__global__ void bucket_hist(const int* __restrict__ y, int* __restrict__ counts, int Bn) {
  int i = blockIdx.x * 256 + threadIdx.x;
  if (i < Bn) atomicAdd(&counts[y[i]], 1);
}
__global__ void bucket_plan(const int* __restrict__ counts, int* __restrict__ cursors,
                            int* __restrict__ tileSpk, int* __restrict__ tileBase,
                            int* __restrict__ tileValid, int* __restrict__ tcnt) {
  if (threadIdx.x == 0 && blockIdx.x == 0) {
    int start = 0, t = 0;
    for (int s = 0; s < NSPK; ++s) {
      int c = counts[s];
      cursors[s] = start;
      int ntl = (c + 63) >> 6;
      for (int i = 0; i < ntl; ++i) {
        int v = c - i * 64; if (v > 64) v = 64;
        tileSpk[t] = s; tileBase[t] = start + i * 64; tileValid[t] = v; ++t;
      }
      start += c;
    }
    *tcnt = t;
  }
}
__global__ void bucket_scatter(const int* __restrict__ y, int* __restrict__ cursors,
                               int* __restrict__ idx, int Bn) {
  int i = blockIdx.x * 256 + threadIdx.x;
  if (i < Bn) { int p = atomicAdd(&cursors[y[i]], 1); idx[p] = i; }
}

// ---------------------------------------------------------------------------
extern "C" void kernel_launch(void* const* d_in, const int* in_sizes, int n_in,
                              void* d_out, int out_size, void* d_ws, size_t ws_size,
                              hipStream_t stream) {
  const float* z   = (const float*)d_in[0];
  const int*   y   = (const int*)  d_in[1];
  const float* sw0 = (const float*)d_in[2];
  const float* sb0 = (const float*)d_in[3];
  const float* sw1 = (const float*)d_in[4];
  const float* sb1 = (const float*)d_in[5];
  const float* sw2 = (const float*)d_in[6];
  const float* sb2 = (const float*)d_in[7];
  const float* sw3 = (const float*)d_in[8];
  const float* sb3 = (const float*)d_in[9];
  const float* uw0 = (const float*)d_in[10];
  const float* ub0 = (const float*)d_in[11];
  const float* uw1 = (const float*)d_in[12];
  const float* ub1 = (const float*)d_in[13];
  const float* uw2 = (const float*)d_in[14];
  const float* ub2 = (const float*)d_in[15];
  const float* uw3 = (const float*)d_in[16];
  const float* ub3 = (const float*)d_in[17];
  float* out = (float*)d_out;

  const int Bn = in_sizes[1];
  const int ntilemax = Bn / 64 + NSPK + 1;

  char* ws = (char*)d_ws;
  size_t o = 0;
  auto carve = [&](size_t bytes) -> char* {
    o = (o + 255) & ~(size_t)255;
    char* p = ws + o; o += bytes; return p;
  };
  _Float16* h16   = (_Float16*)carve((size_t)Bn * HID * 2);
  _Float16* wt0   = (_Float16*)carve(256 * 32 * 2);
  _Float16* wt1   = (_Float16*)carve((size_t)HID * HID * 2);
  _Float16* wt2   = (_Float16*)carve((size_t)HID * HID * 2);
  _Float16* wt3   = (_Float16*)carve((size_t)HID * HID * 2);
  _Float16* wu0t  = (_Float16*)carve((size_t)NSPK * HID * HID * 2);
  _Float16* wu1t  = (_Float16*)carve((size_t)NSPK * HID * HID * 2);
  _Float16* wu2t  = (_Float16*)carve((size_t)NSPK * HID * HID * 2);
  _Float16* wu3t  = (_Float16*)carve((size_t)NSPK * STYLE * HID * 2);
  int* idx        = (int*)carve((size_t)Bn * 4);
  int* counts     = (int*)carve(64);
  int* cursors    = (int*)carve(64);
  int* tileSpk    = (int*)carve((size_t)ntilemax * 4);
  int* tileBase   = (int*)carve((size_t)ntilemax * 4);
  int* tileValid  = (int*)carve((size_t)ntilemax * 4);
  int* tcnt       = (int*)carve(64);

  // Weight conversion (tiny; weights stay L2-resident afterwards)
  conv_wt0<<<(256 * 32 + 255) / 256, 256, 0, stream>>>(sw0, wt0);
  conv_sq <<<(HID * HID + 255) / 256, 256, 0, stream>>>(sw1, wt1, 1);
  conv_sq <<<(HID * HID + 255) / 256, 256, 0, stream>>>(sw2, wt2, 1);
  conv_sq <<<(HID * HID + 255) / 256, 256, 0, stream>>>(sw3, wt3, 1);
  conv_sq <<<(NSPK * HID * HID + 255) / 256, 256, 0, stream>>>(uw0, wu0t, NSPK);
  conv_sq <<<(NSPK * HID * HID + 255) / 256, 256, 0, stream>>>(uw1, wu1t, NSPK);
  conv_sq <<<(NSPK * HID * HID + 255) / 256, 256, 0, stream>>>(uw2, wu2t, NSPK);
  conv_wu3<<<(NSPK * STYLE * HID + 255) / 256, 256, 0, stream>>>(uw3, wu3t);

  // Bucket samples by speaker
  bucket_zero   <<<1, 32, 0, stream>>>(counts);
  bucket_hist   <<<(Bn + 255) / 256, 256, 0, stream>>>(y, counts, Bn);
  bucket_plan   <<<1, 1, 0, stream>>>(counts, cursors, tileSpk, tileBase, tileValid, tcnt);
  bucket_scatter<<<(Bn + 255) / 256, 256, 0, stream>>>(y, cursors, idx, Bn);

  // Main WMMA pipelines
  trunk_kernel <<<(Bn + 63) / 64, TPB, 0, stream>>>(z, wt0, wt1, wt2, wt3,
                                                    sb0, sb1, sb2, sb3, h16, Bn);
  branch_kernel<<<ntilemax, TPB, 0, stream>>>(h16, wu0t, wu1t, wu2t, wu3t,
                                              ub0, ub1, ub2, ub3,
                                              idx, tileSpk, tileBase, tileValid, tcnt, out);
  (void)n_in; (void)out_size; (void)ws_size;
}
